// CloneTrans_24713241821785
// MI455X (gfx1250) — matrix operand order repaired and checked by reading, not compiled, not run on hardware
//
#include <hip/hip_runtime.h>
#include <hip/hip_bf16.h>
#include <math.h>

#define VOCAB 50000
#define D 512
#define H 8
#define DH 64
#define INNER 512
#define Tn 2048
#define Nb 32
#define En 32768
#define CH 256
#define SEQ 9

typedef __attribute__((ext_vector_type(16))) _Float16 v16h;
typedef __attribute__((ext_vector_type(8)))  float    v8f;

// ---------------- positional encodings ----------------
__global__ void pe_kernel(const float* df, const float* bf, float* pe) {
  int idx = blockIdx.x * blockDim.x + threadIdx.x;   // t*D + c
  if (idx >= Tn * D) return;
  int t = idx / D, c = idx % D;
  const float* src = (c < CH) ? df : bf;
  int cc = c & (CH - 1);
  int i = cc >> 1;
  float inv = expf(-((float)(2 * i) / (float)CH) * 9.210340371976184f); // ln(10000)
  float s = src[t] * inv;
  pe[idx] = (cc & 1) ? cosf(s) : sinf(s);
}

// ---------------- GCN degree / norm ----------------
__global__ void deg_init(float* deg) {
  int t = blockIdx.x * blockDim.x + threadIdx.x;
  if (t < Tn) deg[t] = 1.0f;                         // self loop
}
__global__ void deg_scatter(const int* ei, float* deg) {
  int e = blockIdx.x * blockDim.x + threadIdx.x;
  if (e < En) atomicAdd(&deg[ei[En + e]], 1.0f);     // dst row
}
__global__ void deg_fin(float* deg) {
  int t = blockIdx.x * blockDim.x + threadIdx.x;
  if (t < Tn) { float d = deg[t]; deg[t] = d > 0.0f ? rsqrtf(d) : 0.0f; }
}

// ---------------- xw = enc @ gcn_w  (wave per (n,t) row) ----------------
__global__ void xw_kernel(const int* tokens, const float* embed, const float* gw, float* xw) {
  int wave = (blockIdx.x * blockDim.x + threadIdx.x) >> 5;
  int lane = threadIdx.x & 31;
  if (wave >= Nb * Tn) return;
  int n = wave / Tn, t = wave % Tn;
  int tok = tokens[t * Nb + n];
  const float* er = embed + (size_t)tok * D;
  float acc[8] = {0, 0, 0, 0, 0, 0, 0, 0};
  for (int d = lane; d < D; d += 32) {
    float e = er[d];
#pragma unroll
    for (int k = 0; k < 8; k++) acc[k] += e * gw[d * 8 + k];
  }
#pragma unroll
  for (int k = 0; k < 8; k++) {
#pragma unroll
    for (int off = 16; off >= 1; off >>= 1) acc[k] += __shfl_down(acc[k], off, 32);
  }
  if (lane == 0) {
#pragma unroll
    for (int k = 0; k < 8; k++) xw[(size_t)(n * Tn + t) * 8 + k] = acc[k];
  }
}

// ---------------- gcn out init (bias + self-loop) + edge scatter ----------------
__global__ void gout_init(const float* xw, const float* dinv, const float* gb, float* gout) {
  int idx = blockIdx.x * blockDim.x + threadIdx.x;
  if (idx >= Nb * Tn * 8) return;
  int k = idx & 7;
  int t = (idx >> 3) % Tn;
  float di = dinv[t];
  gout[idx] = gb[k] + xw[idx] * di * di;
}
__global__ void gout_scatter(const int* ei, const float* xw, const float* dinv, float* gout) {
  int idx = blockIdx.x * blockDim.x + threadIdx.x;
  if (idx >= En * Nb) return;
  int e = idx / Nb, n = idx % Nb;
  int s = ei[e], d = ei[En + e];
  float norm = dinv[s] * dinv[d];
  const float* xs = xw + (size_t)(n * Tn + s) * 8;
  float* go = gout + (size_t)(n * Tn + d) * 8;
#pragma unroll
  for (int k = 0; k < 8; k++) atomicAdd(&go[k], xs[k] * norm);
}

// ---------------- att = softmax over t, written to d_out ----------------
__global__ void att_softmax(const float* gout, float* att) {
  __shared__ float red[256];
  int nk = blockIdx.x;
  int n = nk >> 3, k = nk & 7;
  int tid = threadIdx.x;
  float m = -INFINITY;
  for (int t = tid; t < Tn; t += 256) m = fmaxf(m, gout[(size_t)(n * Tn + t) * 8 + k]);
  red[tid] = m; __syncthreads();
  for (int s = 128; s > 0; s >>= 1) { if (tid < s) red[tid] = fmaxf(red[tid], red[tid + s]); __syncthreads(); }
  m = red[0]; __syncthreads();
  float ss = 0.0f;
  for (int t = tid; t < Tn; t += 256) ss += expf(gout[(size_t)(n * Tn + t) * 8 + k] - m);
  red[tid] = ss; __syncthreads();
  for (int s = 128; s > 0; s >>= 1) { if (tid < s) red[tid] += red[tid + s]; __syncthreads(); }
  float inv = 1.0f / red[0];
  for (int t = tid; t < Tn; t += 256)
    att[(size_t)(n * 8 + k) * Tn + t] = expf(gout[(size_t)(n * Tn + t) * 8 + k] - m) * inv;
}

// ---------------- sequence init: [code_token, zeros x8] ----------------
__global__ void seq_init(const float* code, float* seq) {
  int idx = blockIdx.x * blockDim.x + threadIdx.x;
  if (idx >= Nb * SEQ * D) return;
  int d = idx % D;
  int r = (idx / D) % SEQ;
  seq[idx] = (r == 0) ? code[d] : 0.0f;
}

// ---------------- lt[n,k,d] = sum_t att[n,k,t]*(embed+pe) into seq rows 1..8 ----------------
__global__ void lt_kernel(const int* tokens, const float* embed, const float* pe,
                          const float* att, float* seq) {
  const int TCHUNKS = 8, TC = Tn / TCHUNKS;  // 256
  int n = blockIdx.x / TCHUNKS;
  int c = blockIdx.x % TCHUNKS;
  int d0 = threadIdx.x * 2;
  float a0[8] = {0, 0, 0, 0, 0, 0, 0, 0};
  float a1[8] = {0, 0, 0, 0, 0, 0, 0, 0};
  int tok = tokens[(c * TC) * Nb + n];
  for (int tt = 0; tt < TC; tt++) {
    int t = c * TC + tt;
    int toknext = (tt + 1 < TC) ? tokens[(t + 1) * Nb + n] : tok;
    // hide gather latency: prefetch next token's embedding row (global_prefetch_b8)
    __builtin_prefetch(&embed[(size_t)toknext * D + d0], 0, 0);
    float e0 = embed[(size_t)tok * D + d0] + pe[(size_t)t * D + d0];
    float e1 = embed[(size_t)tok * D + d0 + 1] + pe[(size_t)t * D + d0 + 1];
#pragma unroll
    for (int k = 0; k < 8; k++) {
      float a = att[(size_t)(n * 8 + k) * Tn + t];
      a0[k] += a * e0;
      a1[k] += a * e1;
    }
    tok = toknext;
  }
#pragma unroll
  for (int k = 0; k < 8; k++) {
    atomicAdd(&seq[(size_t)(n * SEQ + 1 + k) * D + d0], a0[k]);
    atomicAdd(&seq[(size_t)(n * SEQ + 1 + k) * D + d0 + 1], a1[k]);
  }
}

// ---------------- LayerNorm, one wave per 512-wide row ----------------
__global__ void ln_kernel(const float* X, const float* g, const float* b, float* Y, int rows) {
  int wave = (blockIdx.x * blockDim.x + threadIdx.x) >> 5;
  int lane = threadIdx.x & 31;
  if (wave >= rows) return;
  const float* x = X + (size_t)wave * D;
  float v[16];
  float s = 0.0f;
#pragma unroll
  for (int j = 0; j < 16; j++) { v[j] = x[j * 32 + lane]; s += v[j]; }
#pragma unroll
  for (int off = 16; off >= 1; off >>= 1) s += __shfl_xor(s, off, 32);
  float m = s * (1.0f / (float)D);
  float vs = 0.0f;
#pragma unroll
  for (int j = 0; j < 16; j++) { float dd = v[j] - m; vs += dd * dd; }
#pragma unroll
  for (int off = 16; off >= 1; off >>= 1) vs += __shfl_xor(vs, off, 32);
  float r = rsqrtf(vs * (1.0f / (float)D) + 1e-5f);
  float* y = Y + (size_t)wave * D;
#pragma unroll
  for (int j = 0; j < 16; j++) {
    int d = j * 32 + lane;
    y[d] = (v[j] - m) * r * g[d] + b[d];
  }
}

// ---------------- WMMA GEMM with async double-buffered LDS staging ----------------
// C = act(A@W + bias) + res. A: MxK f32 row-major, W: KxN f32 row-major.
// One wave per 16x16 output tile. A (16x32) and B (32x16) tiles are staged into
// LDS via global_load_async_to_lds_b128 (ASYNCcnt), double-buffered: issue tile
// k+1, s_wait_asynccnt 8 -> tile k resident, consume while k+1 DMA is in flight.
#define APAD 36   // padded A row (floats): 144B keeps 16B alignment for b128
#define BPAD 20   // padded B row (floats): 80B keeps 16B alignment for b128
__global__ void gemm_wmma(const float* __restrict__ A, const float* __restrict__ W,
                          const float* __restrict__ bias, const float* __restrict__ res,
                          float* __restrict__ C, int M, int K, int N, int act) {
  __shared__ float lds_a[2][4][16][APAD];
  __shared__ float lds_b[2][4][32][BPAD];
  int wslot = threadIdx.x >> 5;                       // wave within block (0..3)
  int wave = (blockIdx.x * blockDim.x + threadIdx.x) >> 5;
  int lane = threadIdx.x & 31;
  int ntiles_n = N >> 4;
  int tiles = (M >> 4) * ntiles_n;
  if (wave >= tiles) return;
  int mt = wave / ntiles_n;
  int nt = wave % ntiles_n;
  int g = lane >> 4;   // half-wave group
  int l = lane & 15;

  auto stage = [&](int slot, int kb) {
    unsigned abase = (unsigned)(uintptr_t)&lds_a[slot][wslot][0][0];
    unsigned bbase = (unsigned)(uintptr_t)&lds_b[slot][wslot][0][0];
    // A tile: 16 rows x 32 floats = 128 x b128 segments, 4 per lane
#pragma unroll
    for (int j = 0; j < 4; j++) {
      int flat = j * 32 + lane;                  // 0..127
      int row = flat >> 3;                       // 8 segments per row
      int col = (flat & 7) << 2;
      unsigned laddr = abase + (unsigned)((row * APAD + col) * 4);
      const float* gp = A + (size_t)(mt * 16 + row) * K + kb + col;
      asm volatile("global_load_async_to_lds_b128 %0, %1, off"
                   :: "v"(laddr), "v"((unsigned long long)(uintptr_t)gp)
                   : "memory");
    }
    // B tile: 32 rows x 16 floats = 128 x b128 segments, 4 per lane
#pragma unroll
    for (int j = 0; j < 4; j++) {
      int flat = j * 32 + lane;
      int row = flat >> 2;                       // 4 segments per row
      int col = (flat & 3) << 2;
      unsigned laddr = bbase + (unsigned)((row * BPAD + col) * 4);
      const float* gp = W + (size_t)(kb + row) * N + nt * 16 + col;
      asm volatile("global_load_async_to_lds_b128 %0, %1, off"
                   :: "v"(laddr), "v"((unsigned long long)(uintptr_t)gp)
                   : "memory");
    }
  };

  v8f c = {0.f, 0.f, 0.f, 0.f, 0.f, 0.f, 0.f, 0.f};
  stage(0, 0);                                    // prologue: tile 0 in flight
  for (int kb = 0; kb < K; kb += 32) {
    int cur = (kb >> 5) & 1;
    if (kb + 32 < K) {
      stage(cur ^ 1, kb + 32);                    // 8 more async ops in flight
      asm volatile("s_wait_asynccnt 8" ::: "memory");  // oldest 8 (tile cur) done
    } else {
      asm volatile("s_wait_asynccnt 0" ::: "memory");  // drain
    }
    v16h a, b;
    // A layout: lane group g covers K = {16g..16g+15} -> one
    // contiguous 16-float run: read as 4x ds_load_b128 and convert.
    const float* as = &lds_a[cur][wslot][l][g << 4];
    float4 x0 = *(const float4*)(as);
    float4 x1 = *(const float4*)(as + 4);
    float4 x2 = *(const float4*)(as + 8);
    float4 x3 = *(const float4*)(as + 12);
    a[0] = (_Float16)x0.x; a[1] = (_Float16)x0.y; a[2] = (_Float16)x0.z; a[3] = (_Float16)x0.w;
    a[4] = (_Float16)x1.x; a[5] = (_Float16)x1.y; a[6] = (_Float16)x1.z; a[7] = (_Float16)x1.w;
    a[8] = (_Float16)x2.x; a[9] = (_Float16)x2.y; a[10] = (_Float16)x2.z; a[11] = (_Float16)x2.w;
    a[12] = (_Float16)x3.x; a[13] = (_Float16)x3.y; a[14] = (_Float16)x3.z; a[15] = (_Float16)x3.w;
    // B layout: lanes 0-15 hold K=0..15, lanes 16-31 hold K=16..31, column l
#pragma unroll
    for (int h = 0; h < 16; h++) b[h] = (_Float16)lds_b[cur][wslot][h + (g << 4)][l];
    c = __builtin_amdgcn_wmma_f32_16x16x32_f16(false, a, false, b, (short)0, c, false, false);
  }
  int col = nt * 16 + l;
  float bv = bias ? bias[col] : 0.0f;
#pragma unroll
  for (int r = 0; r < 8; r++) {
    int row = mt * 16 + r + (g << 3);
    float v = c[r] + bv;
    if (act) v = 0.5f * v * (1.0f + erff(v * 0.70710678118654752f)); // exact GELU
    if (res) v += res[(size_t)row * N + col];
    C[(size_t)row * N + col] = v;
  }
}

// ---------------- exact top-k attention (kk<=3), thread per (n,h,row) ----------------
__global__ void topk_attn(const float* Q, int qstride, int qoff0,
                          const float* KV, int kvstride, int koff0, int voff0,
                          int kk, float* Out) {
  int idx = blockIdx.x * blockDim.x + threadIdx.x;
  if (idx >= Nb * H * SEQ) return;
  int i = idx % SEQ;
  int nh = idx / SEQ;
  int h = nh % H;
  int n = nh / H;
  const float* q  = Q  + (size_t)(n * SEQ + i) * qstride + qoff0 + h * DH;
  const float* Kb = KV + (size_t)n * SEQ * kvstride + koff0 + h * DH;
  const float* Vb = KV + (size_t)n * SEQ * kvstride + voff0 + h * DH;
  float dots[SEQ];
  const float scale = 0.125f;  // DH^-0.5
  for (int j = 0; j < SEQ; j++) {
    float s = 0.0f;
    const float* kr = Kb + (size_t)j * kvstride;
    for (int d = 0; d < DH; d++) s += q[d] * kr[d];
    dots[j] = s * scale;
  }
  int sel[3]; float vals[3];
  for (int c = 0; c < kk; c++) {          // stable argmax (first index wins ties)
    int bi = 0; float bv = -INFINITY;
    for (int j = 0; j < SEQ; j++) if (dots[j] > bv) { bv = dots[j]; bi = j; }
    sel[c] = bi; vals[c] = bv; dots[bi] = -INFINITY;
  }
  float m = vals[0];
  float w[3]; float se = 0.0f;
  for (int c = 0; c < kk; c++) { w[c] = expf(vals[c] - m); se += w[c]; }
  float inv = 1.0f / se;
  float* o = Out + (size_t)(n * SEQ + i) * D + h * DH;
  for (int d = 0; d < DH; d++) {
    float s = 0.0f;
    for (int c = 0; c < kk; c++) s += w[c] * Vb[(size_t)sel[c] * kvstride + d];
    o[d] = s * inv;
  }
}

// ---------------- gather row 0 of each sequence ----------------
__global__ void head_gather(const float* seq, float* out) {
  int idx = blockIdx.x * blockDim.x + threadIdx.x;
  if (idx >= Nb * D) return;
  int n = idx / D, d = idx % D;
  out[idx] = seq[(size_t)n * SEQ * D + d];
}

// ================= host side =================
static void run_gemm(const float* A, const float* W, const float* bias, const float* res,
                     float* C, int M, int K, int N, int act, hipStream_t st) {
  int tiles = (M / 16) * (N / 16);
  int blocks = (tiles + 3) / 4;                       // 4 waves (128 thr) per block
  gemm_wmma<<<blocks, 128, 0, st>>>(A, W, bias, res, C, M, K, N, act);
}
static void run_ln(const float* X, const float* g, const float* b, float* Y, int rows,
                   hipStream_t st) {
  int blocks = (rows * 32 + 255) / 256;
  ln_kernel<<<blocks, 256, 0, st>>>(X, g, b, Y, rows);
}

extern "C" void kernel_launch(void* const* d_in, const int* in_sizes, int n_in,
                              void* d_out, int out_size, void* d_ws, size_t ws_size,
                              hipStream_t stream) {
  (void)in_sizes; (void)n_in; (void)out_size; (void)ws_size;
  const int*   x_tokens = (const int*)d_in[0];
  const int*   x_ei     = (const int*)d_in[1];
  const float* x_df     = (const float*)d_in[2];
  const float* x_bf     = (const float*)d_in[3];
  const int*   y_tokens = (const int*)d_in[4];
  const int*   y_ei     = (const int*)d_in[5];
  const float* y_df     = (const float*)d_in[6];
  const float* y_bf     = (const float*)d_in[7];
  const float* embed    = (const float*)d_in[8];
  const float* code     = (const float*)d_in[9];
  const float* gcn_w    = (const float*)d_in[10];
  const float* gcn_b    = (const float*)d_in[11];
  const float* mlp_w    = (const float*)d_in[49];
  const float* mlp_b    = (const float*)d_in[50];

  float* ws = (float*)d_ws;
  size_t off = 0;
  auto alloc = [&](size_t nf) { float* p = ws + off; off += nf; return p; };
  float* peX   = alloc((size_t)Tn * D);
  float* peY   = alloc((size_t)Tn * D);
  float* dinvX = alloc(Tn);
  float* dinvY = alloc(Tn);
  float* xwX   = alloc((size_t)Nb * Tn * 8);
  float* xwY   = alloc((size_t)Nb * Tn * 8);
  float* goX   = alloc((size_t)Nb * Tn * 8);
  float* goY   = alloc((size_t)Nb * Tn * 8);
  float* seqX  = alloc((size_t)Nb * SEQ * D);
  float* seqY  = alloc((size_t)Nb * SEQ * D);
  const size_t TB = (size_t)Nb * SEQ * 3 * INNER;     // 442368 floats
  float* bufA = alloc(TB);
  float* bufB = alloc(TB);
  float* bufC = alloc(TB);
  float* bufD = alloc(TB);
  float* bufE = alloc(TB);
  float* bufF = alloc(TB);

  float* out   = (float*)d_out;
  float* xo    = out;
  float* yo    = out + Nb * D;
  float* x_att = out + 2 * Nb * D;
  float* y_att = x_att + (size_t)Nb * H * Tn;

  auto side = [&](const int* tokens, const int* ei, const float* df, const float* bf,
                  float* pe, float* dinv, float* xw, float* go, float* seq, float* att) {
    pe_kernel<<<(Tn * D + 255) / 256, 256, 0, stream>>>(df, bf, pe);
    deg_init<<<(Tn + 255) / 256, 256, 0, stream>>>(dinv);
    deg_scatter<<<(En + 255) / 256, 256, 0, stream>>>(ei, dinv);
    deg_fin<<<(Tn + 255) / 256, 256, 0, stream>>>(dinv);
    xw_kernel<<<(Nb * Tn * 32) / 256, 256, 0, stream>>>(tokens, embed, gcn_w, xw);
    gout_init<<<(Nb * Tn * 8 + 255) / 256, 256, 0, stream>>>(xw, dinv, gcn_b, go);
    gout_scatter<<<(En * Nb + 255) / 256, 256, 0, stream>>>(ei, xw, dinv, go);
    att_softmax<<<Nb * 8, 256, 0, stream>>>(go, att);
    seq_init<<<(Nb * SEQ * D + 255) / 256, 256, 0, stream>>>(code, seq);
    lt_kernel<<<Nb * 8, 256, 0, stream>>>(tokens, embed, pe, att, seq);
  };
  side(x_tokens, x_ei, x_df, x_bf, peX, dinvX, xwX, goX, seqX, x_att);
  side(y_tokens, y_ei, y_df, y_bf, peY, dinvY, xwY, goY, seqY, y_att);

  const int rows = Nb * SEQ;  // 288
  const int ATHREADS = Nb * H * SEQ;

  // ---- 3 shared self layers, applied to both sequences ----
  for (int L = 0; L < 3; L++) {
    const float* ln1g = (const float*)d_in[12 + L * 9 + 0];
    const float* ln1b = (const float*)d_in[12 + L * 9 + 1];
    const float* wqkv = (const float*)d_in[12 + L * 9 + 2];
    const float* wo   = (const float*)d_in[12 + L * 9 + 3];
    const float* bo   = (const float*)d_in[12 + L * 9 + 4];
    const float* ln2g = (const float*)d_in[12 + L * 9 + 5];
    const float* ln2b = (const float*)d_in[12 + L * 9 + 6];
    const float* wff  = (const float*)d_in[12 + L * 9 + 7];
    const float* bff  = (const float*)d_in[12 + L * 9 + 8];
    float* seqs[2] = {seqX, seqY};
    for (int sI = 0; sI < 2; sI++) {
      float* seq = seqs[sI];
      run_ln(seq, ln1g, ln1b, bufA, rows, stream);
      run_gemm(bufA, wqkv, nullptr, nullptr, bufB, rows, D, 3 * INNER, 0, stream);
      topk_attn<<<(ATHREADS + 255) / 256, 256, 0, stream>>>(
          bufB, 3 * INNER, 0, bufB, 3 * INNER, INNER, 2 * INNER, 3, bufC);
      run_gemm(bufC, wo, bo, seq, bufD, rows, INNER, D, 0, stream);   // o + x
      run_ln(bufD, ln2g, ln2b, bufA, rows, stream);
      run_gemm(bufA, wff, bff, bufD, seq, rows, D, D, 1, stream);     // gelu(..)+x
    }
  }

  // ---- inter layer ----
  {
    const float* lng  = (const float*)d_in[39 + 0];
    const float* lnb  = (const float*)d_in[39 + 1];
    const float* wq   = (const float*)d_in[39 + 2];
    const float* wkv  = (const float*)d_in[39 + 3];
    const float* wo   = (const float*)d_in[39 + 4];
    const float* bo   = (const float*)d_in[39 + 5];
    const float* ln2g = (const float*)d_in[39 + 6];
    const float* ln2b = (const float*)d_in[39 + 7];
    const float* wff  = (const float*)d_in[39 + 8];
    const float* bff  = (const float*)d_in[39 + 9];
    run_ln(seqX, lng, lnb, bufA, rows, stream);   // xn
    run_ln(seqY, lng, lnb, bufB, rows, stream);   // yn
    // oy = attn(qy, kx, vx) @ wo + bo
    run_gemm(bufB, wq, nullptr, nullptr, bufC, rows, D, INNER, 0, stream);       // qy
    run_gemm(bufA, wkv, nullptr, nullptr, bufD, rows, D, 2 * INNER, 0, stream);  // kx,vx
    topk_attn<<<(ATHREADS + 255) / 256, 256, 0, stream>>>(
        bufC, INNER, 0, bufD, 2 * INNER, 0, INNER, 2, bufE);
    run_gemm(bufE, wo, bo, nullptr, bufF, rows, INNER, D, 0, stream);            // oy
    // ox = attn(qx, ky, vy) @ wo + bo
    run_gemm(bufA, wq, nullptr, nullptr, bufC, rows, D, INNER, 0, stream);       // qx
    run_gemm(bufB, wkv, nullptr, nullptr, bufD, rows, D, 2 * INNER, 0, stream);  // ky,vy
    topk_attn<<<(ATHREADS + 255) / 256, 256, 0, stream>>>(
        bufC, INNER, 0, bufD, 2 * INNER, 0, INNER, 2, bufE);
    run_gemm(bufE, wo, bo, nullptr, bufC, rows, INNER, D, 0, stream);            // ox
    // x' = gelu(ln2(ox)@wff+bff) + xn ; y' = gelu(ln2(oy)@wff+bff) + yn
    run_ln(bufC, ln2g, ln2b, bufD, rows, stream);
    run_gemm(bufD, wff, bff, bufA, seqX, rows, D, D, 1, stream);
    run_ln(bufF, ln2g, ln2b, bufD, rows, stream);
    run_gemm(bufD, wff, bff, bufB, seqY, rows, D, D, 1, stream);
  }

  // ---- output heads ----
  head_gather<<<(Nb * D + 255) / 256, 256, 0, stream>>>(seqX, bufA);
  run_gemm(bufA, mlp_w, mlp_b, nullptr, xo, Nb, D, D, 0, stream);
  head_gather<<<(Nb * D + 255) / 256, 256, 0, stream>>>(seqY, bufB);
  run_gemm(bufB, mlp_w, mlp_b, nullptr, yo, Nb, D, D, 0, stream);
}
